// MultiEmbodimentActionEncoder_77455440216196
// MI455X (gfx1250) — matrix operand order repaired and checked by reading, not compiled, not run on hardware
//
#include <hip/hip_runtime.h>

typedef __attribute__((ext_vector_type(16))) _Float16 v16h;
typedef __attribute__((ext_vector_type(8)))  _Float16 v8h;
typedef __attribute__((ext_vector_type(2)))  _Float16 v2h;
typedef __attribute__((ext_vector_type(8)))  float    v8f;

#define HID    1024
#define NBATCH 64
#define TSEQ   64

// ---------------------------------------------------------------------------
// A-fragment loaders, CDNA5 WMMA f16 16x16x32 layout (ISA 7.12.2):
// lane L: row m = L&15; halfs 0..7 -> K = kb + 8*(L>=16)+i ; halfs 8..15 -> +16
// ---------------------------------------------------------------------------
__device__ __forceinline__ v16h load_a_f16(const _Float16* __restrict__ A, int lda,
                                           int m, int kb, int half) {
  const _Float16* p = A + (size_t)m * lda + kb + 8 * half;
  v8h c0 = *(const v8h*)(p);
  v8h c1 = *(const v8h*)(p + 16);
  v16h r;
#pragma unroll
  for (int i = 0; i < 8; ++i) { r[i] = c0[i]; r[i + 8] = c1[i]; }
  return r;
}

__device__ __forceinline__ v16h load_a_f32(const float* __restrict__ A, int lda,
                                           int m, int kb, int half) {
  const float* p = A + (size_t)m * lda + kb + 8 * half;  // 16B aligned
  float4 q0 = *(const float4*)(p);
  float4 q1 = *(const float4*)(p + 4);
  float4 q2 = *(const float4*)(p + 16);
  float4 q3 = *(const float4*)(p + 20);
  v16h r;
  r[0]  = (_Float16)q0.x; r[1]  = (_Float16)q0.y;
  r[2]  = (_Float16)q0.z; r[3]  = (_Float16)q0.w;
  r[4]  = (_Float16)q1.x; r[5]  = (_Float16)q1.y;
  r[6]  = (_Float16)q1.z; r[7]  = (_Float16)q1.w;
  r[8]  = (_Float16)q2.x; r[9]  = (_Float16)q2.y;
  r[10] = (_Float16)q2.z; r[11] = (_Float16)q2.w;
  r[12] = (_Float16)q3.x; r[13] = (_Float16)q3.y;
  r[14] = (_Float16)q3.z; r[15] = (_Float16)q3.w;
  return r;
}

// D/C epilogue: element (r, lane) -> (m0 + r + 8*(L>=16), n0 + (L&15))
template<bool SWISH, bool OUT_F32>
__device__ __forceinline__ void epilogue_tile(const v8f& acc, int m0, int n0,
                                              int half, int l15,
                                              const float* __restrict__ bias,
                                              void* __restrict__ Out, int ldo) {
#pragma unroll
  for (int r = 0; r < 8; ++r) {
    int m = m0 + r + 8 * half;
    int n = n0 + l15;
    float v = acc[r] + bias[n];
    if (SWISH) v = v * (1.0f / (1.0f + __expf(-v)));
    if (OUT_F32) {
      ((float*)Out)[(size_t)m * ldo + n] = v;
    } else {
      ((_Float16*)Out)[(size_t)m * ldo + n] = (_Float16)v;
    }
  }
}

// ---------------------------------------------------------------------------
// Per-category GEMM with LDS-transposed f16 weight staging.
//   out[b] = act(A[b] @ W[cat_ids[b]] + bias[cat_ids[b]])
// M = 64 rows/batch, N = 1024 cols. WG = 256 thr (8 waves): 64x128 block,
// each wave a 32x32 tile (4 accumulators).
// LDS tile: sW[n][k], n in [0,128), k in [0,KTILE), row stride KTILE+8 halfs
// (16B-aligned rows, staggered banks). B-fragment = two 16B LDS reads (k and k+16).
// grid = (N/128, B), block = 256.
// ---------------------------------------------------------------------------
template<int KTILE, bool A_F16, bool SWISH, bool OUT_F32>
__global__ __launch_bounds__(256)
void cat_gemm_lds(const void* __restrict__ Aall, int lda, long astride,
                  const float* __restrict__ Wall, int K,
                  const float* __restrict__ ball,
                  const int* __restrict__ cat_ids,
                  void* __restrict__ Outall, int ldo, long ostride) {
  constexpr int RS = KTILE + 8;          // LDS row stride in halfs
  __shared__ _Float16 sW[128 * RS];

  const int b   = blockIdx.y;
  const int n0  = blockIdx.x * 128;
  const int tid = threadIdx.x;
  const int wid = tid >> 5;
  const int lid = tid & 31;
  const int half = lid >> 4;
  const int l15  = lid & 15;

  const int mw = (wid & 1) * 32;         // wave M base (0 or 32)
  const int nw = (wid >> 1) * 32;        // wave N base within 128 block

  const int cat = cat_ids[b];
  const float* Wg   = Wall + (size_t)cat * K * HID;
  const float* bias = ball + (size_t)cat * HID;

  v8f acc00 = {}, acc01 = {}, acc10 = {}, acc11 = {};

  for (int kb = 0; kb < K; kb += KTILE) {
    __syncthreads();
    // ---- stage: W[kb..kb+KTILE) x [n0..n0+128) -> sW[n][k] as f16 ----
    // thread handles (k-pair kp, 4-col group n4): two float4 row reads
    // (coalesced), 4 packed cvt, 4 ds_store_b32 into rows 4*n4..4*n4+3.
#pragma unroll
    for (int p = 0; p < KTILE / 16; ++p) {
      int idx = tid + 256 * p;
      int n4  = idx & 31;                // 4-col group, covers 128 cols
      int kp  = idx >> 5;                // k-pair index, 0..KTILE/2-1
      const float* r0p = &Wg[(size_t)(kb + 2 * kp)     * HID + n0 + 4 * n4];
      const float* r1p = &Wg[(size_t)(kb + 2 * kp + 1) * HID + n0 + 4 * n4];
      float4 r0 = *(const float4*)r0p;
      float4 r1 = *(const float4*)r1p;
      int nb = 4 * n4;
      v2h pk0; pk0.x = (_Float16)r0.x; pk0.y = (_Float16)r1.x;
      v2h pk1; pk1.x = (_Float16)r0.y; pk1.y = (_Float16)r1.y;
      v2h pk2; pk2.x = (_Float16)r0.z; pk2.y = (_Float16)r1.z;
      v2h pk3; pk3.x = (_Float16)r0.w; pk3.y = (_Float16)r1.w;
      *(v2h*)&sW[(nb + 0) * RS + 2 * kp] = pk0;
      *(v2h*)&sW[(nb + 1) * RS + 2 * kp] = pk1;
      *(v2h*)&sW[(nb + 2) * RS + 2 * kp] = pk2;
      *(v2h*)&sW[(nb + 3) * RS + 2 * kp] = pk3;
    }
    __syncthreads();

    // ---- compute: 4 WMMAs per 32-wide K step ----
#pragma unroll
    for (int ks = 0; ks < KTILE; ks += 32) {
      v16h a0, a1;
      if constexpr (A_F16) {
        const _Float16* A = (const _Float16*)Aall + (size_t)b * astride;
        a0 = load_a_f16(A, lda, mw + l15,      kb + ks, half);
        a1 = load_a_f16(A, lda, mw + 16 + l15, kb + ks, half);
      } else {
        const float* A = (const float*)Aall + (size_t)b * astride;
        a0 = load_a_f32(A, lda, mw + l15,      kb + ks, half);
        a1 = load_a_f32(A, lda, mw + 16 + l15, kb + ks, half);
      }
      // B-fragment: lane L -> col n = l15(+16), halfs 0..7 = K 8*half + [0,8), halfs 8..15 = +16
      const _Float16* pb0 = &sW[(nw + l15)      * RS + ks + 8 * half];
      const _Float16* pb1 = &sW[(nw + 16 + l15) * RS + ks + 8 * half];
      v16h bf0 = __builtin_shufflevector(*(const v8h*)pb0, *(const v8h*)(pb0 + 16), 0, 1, 2, 3, 4, 5, 6, 7, 8, 9, 10, 11, 12, 13, 14, 15);
      v16h bf1 = __builtin_shufflevector(*(const v8h*)pb1, *(const v8h*)(pb1 + 16), 0, 1, 2, 3, 4, 5, 6, 7, 8, 9, 10, 11, 12, 13, 14, 15);

      acc00 = __builtin_amdgcn_wmma_f32_16x16x32_f16(false, a0, false, bf0, (short)0, acc00, false, false);
      acc01 = __builtin_amdgcn_wmma_f32_16x16x32_f16(false, a0, false, bf1, (short)0, acc01, false, false);
      acc10 = __builtin_amdgcn_wmma_f32_16x16x32_f16(false, a1, false, bf0, (short)0, acc10, false, false);
      acc11 = __builtin_amdgcn_wmma_f32_16x16x32_f16(false, a1, false, bf1, (short)0, acc11, false, false);
    }
  }

  void* Out = OUT_F32 ? (void*)((float*)Outall + (size_t)b * ostride)
                      : (void*)((_Float16*)Outall + (size_t)b * ostride);
  epilogue_tile<SWISH, OUT_F32>(acc00, mw,      n0 + nw,      half, l15, bias, Out, ldo);
  epilogue_tile<SWISH, OUT_F32>(acc01, mw,      n0 + nw + 16, half, l15, bias, Out, ldo);
  epilogue_tile<SWISH, OUT_F32>(acc10, mw + 16, n0 + nw,      half, l15, bias, Out, ldo);
  epilogue_tile<SWISH, OUT_F32>(acc11, mw + 16, n0 + nw + 16, half, l15, bias, Out, ldo);
}

// ---------------------------------------------------------------------------
// Sinusoidal time embedding: tau[b][n] constant over T; fill x[b][t][1024+n].
// grid = (4, B), block = 256.
// ---------------------------------------------------------------------------
__global__ __launch_bounds__(256)
void tau_fill_kernel(const int* __restrict__ timesteps, _Float16* __restrict__ x) {
  const int b = blockIdx.y;
  const int n = blockIdx.x * 256 + threadIdx.x;  // 0..1023
  const float t = (float)timesteps[b];
  const int j = n & 511;
  const float kLog1e4Over512 = 9.210340371976184f / 512.0f;
  const float f = t * __expf(-(float)j * kLog1e4Over512);
  const float v = (n < 512) ? sinf(f) : cosf(f);
  const _Float16 hv = (_Float16)v;
  _Float16* base = x + (size_t)b * TSEQ * (2 * HID) + HID + n;
#pragma unroll 4
  for (int tt = 0; tt < TSEQ; ++tt) base[(size_t)tt * (2 * HID)] = hv;
}

// ---------------------------------------------------------------------------
// Launcher
// ---------------------------------------------------------------------------
extern "C" void kernel_launch(void* const* d_in, const int* in_sizes, int n_in,
                              void* d_out, int out_size, void* d_ws, size_t ws_size,
                              hipStream_t stream) {
  const float* actions   = (const float*)d_in[0];  // (64,64,64) f32
  const int*   timesteps = (const int*)d_in[1];    // (64,) i32
  const int*   cat_ids   = (const int*)d_in[2];    // (64,) i32
  const float* W1        = (const float*)d_in[3];  // (16,64,1024)
  const float* b1        = (const float*)d_in[4];  // (16,1024)
  const float* W2        = (const float*)d_in[5];  // (16,2048,1024)
  const float* b2        = (const float*)d_in[6];  // (16,1024)
  const float* W3        = (const float*)d_in[7];  // (16,1024,1024)
  const float* b3        = (const float*)d_in[8];  // (16,1024)

  // Workspace: x (B,T,2H) f16 = 16 MB, h (B,T,H) f16 = 8 MB
  _Float16* x = (_Float16*)d_ws;
  _Float16* h = (_Float16*)((char*)d_ws + (size_t)NBATCH * TSEQ * (2 * HID) * sizeof(_Float16));

  // 1) sinusoidal half of x
  tau_fill_kernel<<<dim3(HID / 256, NBATCH), 256, 0, stream>>>(timesteps, x);

  dim3 grid(HID / 128, NBATCH);

  // 2) a_emb = actions @ W1[cat] + b1[cat]  ->  x[:, :, 0:1024] (f16)
  cat_gemm_lds<64, false, false, false><<<grid, 256, 0, stream>>>(
      (const void*)actions, 64, (long)TSEQ * 64,
      W1, 64, b1, cat_ids,
      (void*)x, 2 * HID, (long)TSEQ * (2 * HID));

  // 3) h = swish(x @ W2[cat] + b2[cat])  (f16)
  cat_gemm_lds<128, true, true, false><<<grid, 256, 0, stream>>>(
      (const void*)x, 2 * HID, (long)TSEQ * (2 * HID),
      W2, 2 * HID, b2, cat_ids,
      (void*)h, HID, (long)TSEQ * HID);

  // 4) out = h @ W3[cat] + b3[cat]  (f32)
  cat_gemm_lds<128, true, false, true><<<grid, 256, 0, stream>>>(
      (const void*)h, HID, (long)TSEQ * HID,
      W3, HID, b3, cat_ids,
      d_out, HID, (long)TSEQ * HID);
}
